// GraphNet_37701222924405
// MI455X (gfx1250) — compile-verified, matrix operand order checked by reading, not run on hardware
//
#include <hip/hip_runtime.h>
#include <hip/hip_bf16.h>
#include <stdint.h>
#include <stddef.h>

// ---------------------------------------------------------------------------
// MEGNet-style GraphNet for MI455X (gfx1250, wave32, WMMA).
// Heavy GEMMs run as bf16 WMMA (v_wmma_f32_16x16x32_bf16) with f32 accum.
// Weights live in LDS (NSEG*32KB of the 320 KB/WGP), pre-swizzled into WMMA
// B-fragment layout so each fragment load is a single pair of ds_load_b128.
// Kernel is templated on the segment count so the K-loop is branch-free.
// ---------------------------------------------------------------------------

typedef __attribute__((ext_vector_type(16))) __bf16 v16bf;
typedef __attribute__((ext_vector_type(8)))  float  v8f;

#define DD       128
#define N_ATOMS_ 65536
#define N_BONDS_ 131072
#define N_MOLS_  512

__device__ __forceinline__ unsigned short f2bf(float f) {
  union { float f; unsigned u; } x; x.f = f;
  unsigned r = x.u + 0x7FFFu + ((x.u >> 16) & 1u);   // round-to-nearest-even
  return (unsigned short)(r >> 16);
}

__device__ __forceinline__ float relu_(float v) { return v > 0.f ? v : 0.f; }

// ------------------------------- small kernels ------------------------------

__global__ void k_zero(float* p, long n) {
  long i = (long)blockIdx.x * blockDim.x + threadIdx.x;
  if (i < n) p[i] = 0.f;
}

__global__ void k_copy(const float* s, float* d, int n) {
  int i = blockIdx.x * blockDim.x + threadIdx.x;
  if (i < n) d[i] = s[i];
}

// h_v0 = relu(atoms @ W_fv + b_fv);  also seed h_v = h_v0
__global__ void k_init_hv0(const float* __restrict__ atoms,
                           const float* __restrict__ W_fv,
                           const float* __restrict__ b_fv,
                           float* __restrict__ h_v0, float* __restrict__ h_v) {
  int id  = blockIdx.x * blockDim.x + threadIdx.x;   // N_ATOMS * 128 threads
  int col = id & (DD - 1);
  int a   = id >> 7;
  const float* ar = atoms + (size_t)a * 16;
  float acc = b_fv[col];
#pragma unroll
  for (int k = 0; k < 16; ++k) acc += ar[k] * W_fv[k * DD + col];
  float v = relu_(acc);
  h_v0[id] = v;
  h_v[id]  = v;
}

// h_e seeded with h_e0 = relu(bo * W_fe + b_fe)  (rank-1)
__global__ void k_init_he(const float* __restrict__ bo,
                          const float* __restrict__ wfe,
                          const float* __restrict__ bfe,
                          float* __restrict__ h_e) {
  int id  = blockIdx.x * blockDim.x + threadIdx.x;   // N_BONDS * 128 threads
  int col = id & (DD - 1);
  int e   = id >> 7;
  h_e[id] = relu_(bo[e] * wfe[col] + bfe[col]);
}

// segment-sum rows of x[n_rows,128] into outm[.,128] by sorted row->segment
// ids; 8 rows per thread with run-accumulation to cut atomic traffic ~8x.
__global__ void k_scatter_rows(const float* __restrict__ x,
                               const int* __restrict__ rid,
                               float* __restrict__ outm, int n_rows) {
  int id  = blockIdx.x * blockDim.x + threadIdx.x;   // (n_rows/8)*128 threads
  int col = id & (DD - 1);
  int r0  = (id >> 7) * 8;
  if (r0 >= n_rows) return;
  float acc = 0.f;
  int cur = rid[r0];
#pragma unroll
  for (int i = 0; i < 8; ++i) {
    int r = r0 + i;
    float v = x[(size_t)r * DD + col];
    int m = rid[r];
    if (m != cur) { atomicAdd(&outm[cur * DD + col], acc); acc = 0.f; cur = m; }
    acc += v;
  }
  atomicAdd(&outm[cur * DD + col], acc);
}

// e_bar_i[src]+=h_e, e_bar_i[dst]+=h_e (random targets -> direct atomics) and
// e_bar[bond_mol]+=h_e (sorted -> run-accumulate).
__global__ void k_scatter_edges(const float* __restrict__ h_e,
                                const int* __restrict__ src,
                                const int* __restrict__ dst,
                                const int* __restrict__ bmol,
                                float* __restrict__ e_bar_i,
                                float* __restrict__ e_bar) {
  int id  = blockIdx.x * blockDim.x + threadIdx.x;   // (E/8)*128 threads
  int col = id & (DD - 1);
  int e0  = (id >> 7) * 8;
  if (e0 >= N_BONDS_) return;
  float acc = 0.f;
  int cur = bmol[e0];
#pragma unroll
  for (int i = 0; i < 8; ++i) {
    int e = e0 + i;
    float v = h_e[(size_t)e * DD + col];
    atomicAdd(&e_bar_i[(size_t)src[e] * DD + col], v);
    atomicAdd(&e_bar_i[(size_t)dst[e] * DD + col], v);
    int m = bmol[e];
    if (m != cur) { atomicAdd(&e_bar[cur * DD + col], acc); acc = 0.f; cur = m; }
    acc += v;
  }
  atomicAdd(&e_bar[cur * DD + col], acc);
}

// ------------------------- fused concat-GEMM (WMMA) -------------------------
// out[r, :] = relu( concat(seg_0[r],...,seg_{NSEG-1}[r]) @ W + bias )
// Each 128-wide segment is (base[idx[r]]) or, if base==nullptr, the rank-1
// recompute relu(bo[r]*wfe + bfe) (used for h_e0 to avoid materializing 64MB).
// Block: 256 threads = 8 waves; each wave computes 2 sequential strips of
// 16 rows x 128 cols (8 C-tiles each) -> block covers 256 rows, amortizing
// the LDS weight staging twice as far. K loop: NSEG*4 steps of 32.
// W (fp32 global) is converted to bf16 and swizzled into LDS in B-fragment
// order: frag(step s, col-tile t) = 32 lanes x 16 bf16, contiguous 32B/lane.
//   A fragment (16-bit 16x32, ISA 7.12.2): lane<16 -> row=lane,   k in {kk+0..7, kk+16..23}
//                                          lane>=16 -> row=lane-16,k in {kk+8..15,kk+24..31}
//   B fragment mirrors with columns;  C/D: vgpr j = row (lane<16 ? j : j+8), col = t*16+(lane&15)
template <int NSEG>
__launch_bounds__(256)
__global__ void k_concat_gemm(
    const float* s0, const int* i0,
    const float* s1, const int* i1,
    const float* s2, const int* i2,
    const float* s3, const int* i3,
    const float* s4, const int* i4,
    const float* __restrict__ bo,      // rank-1 segment inputs (base==nullptr)
    const float* __restrict__ wfe,
    const float* __restrict__ bfe,
    const float* __restrict__ W,       // [NSEG*128, 128] fp32 row-major
    const float* __restrict__ bias,    // [128]
    float* out)                        // [gridDim.x*256, 128], in-place OK
{
  __shared__ alignas(32) unsigned short ldsW[NSEG * DD * DD];  // NSEG*32 KB

  const int tid  = threadIdx.x;
  const int lane = tid & 31;
  const int wave = tid >> 5;

  // Cooperative fp32->bf16 convert + fragment swizzle of W into LDS.
  for (int id = tid; id < NSEG * DD * DD; id += 256) {
    int k = id >> 7, n = id & (DD - 1);
    int s = k >> 5, kr = k & 31;
    int l, j;
    if      (kr <  8) { l = (n & 15);      j = kr;      }
    else if (kr < 16) { l = 16 + (n & 15); j = kr - 8;  }
    else if (kr < 24) { l = (n & 15);      j = kr - 8;  }
    else              { l = 16 + (n & 15); j = kr - 16; }
    int t = n >> 4;
    ldsW[(((s * 8 + t) * 32) + l) * 16 + j] = f2bf(W[id]);
  }
  __syncthreads();

  const int kshift = (lane < 16) ? 0 : 8;

#pragma unroll 1
  for (int strip = 0; strip < 2; ++strip) {
    const int row_base = blockIdx.x * 256 + (strip * 8 + wave) * 16;
    const int row      = row_base + (lane & 15);

    v8f c[8] = {};

#pragma unroll
    for (int seg = 0; seg < NSEG; ++seg) {
      const float* base; const int* idx;
      switch (seg) {
        case 0:  base = s0; idx = i0; break;
        case 1:  base = s1; idx = i1; break;
        case 2:  base = s2; idx = i2; break;
        case 3:  base = s3; idx = i3; break;
        default: base = s4; idx = i4; break;
      }
      const int r = idx ? idx[row] : row;
      float bov = 0.f;
      if (!base) bov = bo[row];

#pragma unroll
      for (int ks = 0; ks < 4; ++ks) {
        const int kk = ks * 32 + kshift;   // this lane's first k within segment
        float vals[16];
        if (base) {
          const float* p = base + (size_t)r * DD + kk;
#pragma unroll
          for (int j = 0; j < 8; ++j) vals[j] = p[j];
#pragma unroll
          for (int j = 0; j < 8; ++j) vals[8 + j] = p[16 + j];
        } else {
#pragma unroll
          for (int j = 0; j < 8; ++j) vals[j] = relu_(bov * wfe[kk + j] + bfe[kk + j]);
#pragma unroll
          for (int j = 0; j < 8; ++j) vals[8 + j] = relu_(bov * wfe[kk + 16 + j] + bfe[kk + 16 + j]);
        }
        v16bf a;
#pragma unroll
        for (int j = 0; j < 16; ++j) a[j] = (__bf16)vals[j];

        const int sstep = seg * 4 + ks;
#pragma unroll
        for (int t = 0; t < 8; ++t) {
          v16bf b = *(const v16bf*)&ldsW[((sstep * 8 + t) * 32 + lane) * 16];
          c[t] = __builtin_amdgcn_wmma_f32_16x16x32_bf16(
              /*neg_a=*/false, a, /*neg_b=*/false, b,
              /*c_mod=*/(short)0, c[t], /*reuse_a=*/false, /*reuse_b=*/false);
        }
      }
    }

    // Epilogue: bias + relu, fp32 store (C/D layout per ISA 7.12.2).
    const int col0 = lane & 15;
    const int rb   = row_base + ((lane < 16) ? 0 : 8);
#pragma unroll
    for (int t = 0; t < 8; ++t) {
      const int col = t * 16 + col0;
      const float bv = bias[col];
#pragma unroll
      for (int j = 0; j < 8; ++j) {
        float v = c[t][j] + bv;
        out[(size_t)(rb + j) * DD + col] = relu_(v);
      }
    }
  }
}

// ------------------------------- host driver --------------------------------

extern "C" void kernel_launch(void* const* d_in, const int* in_sizes, int n_in,
                              void* d_out, int out_size, void* d_ws, size_t ws_size,
                              hipStream_t stream) {
  const float* atoms       = (const float*)d_in[0];
  const float* bond_orders = (const float*)d_in[1];
  const float* W_fe = (const float*)d_in[2];
  const float* b_fe = (const float*)d_in[3];
  const float* W_fv = (const float*)d_in[4];
  const float* b_fv = (const float*)d_in[5];
  const float* W_fu = (const float*)d_in[6];
  const float* b_fu = (const float*)d_in[7];
  const float* W_e  = (const float*)d_in[8];
  const float* b_e  = (const float*)d_in[9];
  const float* W_v  = (const float*)d_in[10];
  const float* b_v  = (const float*)d_in[11];
  const float* W_u  = (const float*)d_in[12];
  const float* b_u  = (const float*)d_in[13];
  const int* bond_src = (const int*)d_in[14];
  const int* bond_dst = (const int*)d_in[15];
  const int* atom_mol = (const int*)d_in[16];
  const int* bond_mol = (const int*)d_in[17];

  // workspace layout (floats): ~161 MB total
  float* ws      = (float*)d_ws;
  float* h_v0    = ws;                                   // N*128
  float* h_v     = h_v0    + (size_t)N_ATOMS_ * DD;      // N*128
  float* h_e     = h_v     + (size_t)N_ATOMS_ * DD;      // E*128
  float* e_bar_i = h_e     + (size_t)N_BONDS_ * DD;      // N*128
  float* e_bar   = e_bar_i + (size_t)N_ATOMS_ * DD;      // M*128
  float* v_bar   = e_bar   + (size_t)N_MOLS_  * DD;      // M*128
  float* s_v0    = v_bar   + (size_t)N_MOLS_  * DD;      // M*128
  float* h_u0    = s_v0    + (size_t)N_MOLS_  * DD;      // M*128
  float* h_u     = (float*)d_out;                        // M*128, updated in place

  const int B = 256;
  (void)in_sizes; (void)n_in; (void)out_size; (void)ws_size;

  // ---- initial embeddings ----
  k_init_hv0<<<(N_ATOMS_ * DD) / B, B, 0, stream>>>(atoms, W_fv, b_fv, h_v0, h_v);
  k_init_he <<<(N_BONDS_ * DD) / B, B, 0, stream>>>(bond_orders, W_fe, b_fe, h_e);

  k_zero<<<(N_MOLS_ * DD + B - 1) / B, B, 0, stream>>>(s_v0, (long)N_MOLS_ * DD);
  k_scatter_rows<<<((N_ATOMS_ / 8) * DD) / B, B, 0, stream>>>(h_v0, atom_mol, s_v0, N_ATOMS_);

  // h_u0 = relu(s_v0 @ W_fu + b_fu)   (512 rows -> 2 blocks)
  k_concat_gemm<1><<<N_MOLS_ / 256, B, 0, stream>>>(
      s_v0, nullptr, nullptr, nullptr, nullptr, nullptr, nullptr, nullptr, nullptr, nullptr,
      nullptr, nullptr, nullptr,
      W_fu, b_fu, h_u0);
  k_copy<<<(N_MOLS_ * DD) / B, B, 0, stream>>>(h_u0, h_u, N_MOLS_ * DD);

  // ---- 3 message-passing iterations ----
  for (int it = 0; it < 3; ++it) {
    // phi_e: concat(h_e, h_e0(rank-1), h_v[src], h_v[dst], h_u[bond_mol]) @ W_e
    k_concat_gemm<5><<<N_BONDS_ / 256, B, 0, stream>>>(
        h_e, nullptr,
        nullptr, nullptr,            // rank-1 h_e0, recomputed on the fly
        h_v, bond_src,
        h_v, bond_dst,
        h_u, bond_mol,
        bond_orders, W_fe, b_fe,
        W_e, b_e, h_e);

    // zero e_bar_i | e_bar | v_bar (contiguous)
    k_zero<<<((long)(N_ATOMS_ * DD + 2 * N_MOLS_ * DD) + B - 1) / B, B, 0, stream>>>(
        e_bar_i, (long)N_ATOMS_ * DD + 2L * N_MOLS_ * DD);

    k_scatter_edges<<<((N_BONDS_ / 8) * DD) / B, B, 0, stream>>>(
        h_e, bond_src, bond_dst, bond_mol, e_bar_i, e_bar);

    // phi_v: concat(h_v, h_v0, e_bar_i, h_u[atom_mol]) @ W_v
    k_concat_gemm<4><<<N_ATOMS_ / 256, B, 0, stream>>>(
        h_v, nullptr,
        h_v0, nullptr,
        e_bar_i, nullptr,
        h_u, atom_mol,
        nullptr, nullptr,
        nullptr, nullptr, nullptr,
        W_v, b_v, h_v);

    k_scatter_rows<<<((N_ATOMS_ / 8) * DD) / B, B, 0, stream>>>(h_v, atom_mol, v_bar, N_ATOMS_);

    // phi_u: concat(h_u, h_u0, e_bar, v_bar) @ W_u   (in place on d_out)
    k_concat_gemm<4><<<N_MOLS_ / 256, B, 0, stream>>>(
        h_u, nullptr,
        h_u0, nullptr,
        e_bar, nullptr,
        v_bar, nullptr,
        nullptr, nullptr,
        nullptr, nullptr, nullptr,
        W_u, b_u, h_u);
  }
}
